// HGNNLayer_8967891714518
// MI455X (gfx1250) — matrix operand order repaired and checked by reading, not compiled, not run on hardware
//
#include <hip/hip_runtime.h>

// ============================================================================
// HGNN layer for MI455X (gfx1250, wave32, WMMA).
//
// out = x @ W_c^T + b_c                       (dense, 13 GFLOP)
//     + scatter_norm(gather(x) @ A_s1[type])  (shape-1 edges, 39 GFLOP)
//     + scatter_norm(gather(x) @ A_s2[type])  (shape-2 edges, 79 GFLOP)
//
// All three GEMMs run on v_wmma_f32_16x16x32_bf16 with a hi/lo bf16 split
// (hi*hi + hi*lo + lo*hi) for ~fp32 accuracy. 16 rows (edges/nodes) form the
// M dimension per workgroup; 8 waves x 2 N-tiles cover the 256 output cols.
// B operands are pre-swizzled into fragment order (coalesced b128 loads from
// L2); gathered x rows are split fp32->bf16(hi,lo) into LDS. Scatter uses
// native fp32 global atomics (relaxed, agent scope).
// ============================================================================

#define DDIM 256

typedef __attribute__((ext_vector_type(16))) __bf16 v16bf;
typedef __attribute__((ext_vector_type(8)))  float  v8f;

union BFrag {
  uint4 q[2];
  v16bf v;
};

__device__ __forceinline__ unsigned short f2bf(float f) {
  unsigned int u = __float_as_uint(f);
  u += 0x7FFFu + ((u >> 16) & 1u);          // round-to-nearest-even
  return (unsigned short)(u >> 16);
}
__device__ __forceinline__ float bf2f(unsigned short h) {
  return __uint_as_float(((unsigned int)h) << 16);
}
// split two fp32 into packed bf16 hi-plane / lo-plane dwords
__device__ __forceinline__ void split_pack(float f0, float f1,
                                           unsigned int& hi, unsigned int& lo) {
  unsigned short h0 = f2bf(f0); unsigned short l0 = f2bf(f0 - bf2f(h0));
  unsigned short h1 = f2bf(f1); unsigned short l1 = f2bf(f1 - bf2f(h1));
  hi = (unsigned int)h0 | ((unsigned int)h1 << 16);
  lo = (unsigned int)l0 | ((unsigned int)l1 << 16);
}

__device__ __forceinline__ v8f wmma_bf16(const BFrag& a, const BFrag& b, v8f c) {
  return __builtin_amdgcn_wmma_f32_16x16x32_bf16(
      /*neg_a=*/false, a.v, /*neg_b=*/false, b.v,
      /*c_mod=*/(short)0, c, /*reuse_a=*/false, /*reuse_b=*/false);
}

// native fp32 atomic add (relaxed, device scope) -> global_atomic_add_f32
__device__ __forceinline__ void atomic_fadd(float* p, float v) {
  __hip_atomic_fetch_add(p, v, __ATOMIC_RELAXED, __HIP_MEMORY_SCOPE_AGENT);
}
__device__ __forceinline__ int atomic_iadd(int* p, int v) {
  return __hip_atomic_fetch_add(p, v, __ATOMIC_RELAXED, __HIP_MEMORY_SCOPE_AGENT);
}

// ---------------------------------------------------------------------------
// Pre-swizzle weights into WMMA B-fragment order, split into bf16 hi/lo.
// Fragment dword index: (((t*KT + kt)*16 + ntile)*32 + lane)*8 + v
// holds pack(B[k][n], B[k+1][n]), k = kt*32 + (v/4)*16 + (lane/16)*8 + 2*(v%4),
// n = ntile*16 + (lane%16).  TRANSPOSE=true reads B[k][n] = src[n][k] (W_c).
// ---------------------------------------------------------------------------
template <int KDIM, bool TRANSPOSE>
__global__ void hgnn_swizzleB(const float* __restrict__ src,
                              unsigned int* __restrict__ hi,
                              unsigned int* __restrict__ lo, int T) {
  constexpr int KT = KDIM / 32;
  int gid = blockIdx.x * blockDim.x + threadIdx.x;
  int total = T * KT * 16 * 32 * 8;
  if (gid >= total) return;
  int v     = gid & 7;
  int lane  = (gid >> 3) & 31;
  int ntile = (gid >> 8) & 15;
  int rest  = gid >> 12;
  int kt    = rest % KT;
  int t     = rest / KT;
  int lg = lane >> 4;
  int n  = ntile * 16 + (lane & 15);
  int k  = kt * 32 + (v >> 2) * 16 + lg * 8 + 2 * (v & 3);
  float f0, f1;
  if (TRANSPOSE) {
    f0 = src[(size_t)n * DDIM + k];
    f1 = src[(size_t)n * DDIM + k + 1];
  } else {
    f0 = src[((size_t)t * KDIM + k) * DDIM + n];
    f1 = src[((size_t)t * KDIM + k + 1) * DDIM + n];
  }
  unsigned int h, l;
  split_pack(f0, f1, h, l);
  hi[gid] = h;
  lo[gid] = l;
}

// ---------------------------------------------------------------------------
// Per-edge pass: count edges per (type, dst) and bucket edge ids by type.
// ---------------------------------------------------------------------------
__global__ void hgnn_count_bucket(const int* __restrict__ idx,
                                  const int* __restrict__ etype,
                                  int* __restrict__ counts,
                                  int* __restrict__ lens,
                                  int* __restrict__ lists,
                                  int E, int N, int shape) {
  int e = blockIdx.x * blockDim.x + threadIdx.x;
  if (e >= E) return;
  int t   = etype[e];
  int dst = idx[(size_t)E * shape + (size_t)shape * e];  // idx1[shape*e]
  atomic_iadd(&counts[(size_t)t * N + dst], 1);
  int p = atomic_iadd(&lens[t], 1);
  lists[(size_t)t * E + p] = e;
}

// ---------------------------------------------------------------------------
// Dense: out = x @ W_c^T + b_c   (writes out, runs before edge kernels)
// ---------------------------------------------------------------------------
__global__ __launch_bounds__(256) void hgnn_dense_wmma(
    const float* __restrict__ x, const unsigned int* __restrict__ WHi,
    const unsigned int* __restrict__ WLo, const float* __restrict__ bc,
    float* __restrict__ out, int N) {
  constexpr int KT = DDIM / 32;        // 8
  constexpr int PITCH = DDIM / 2 + 4;  // dwords, pad for LDS bank spread
  __shared__ __align__(16) unsigned int sHi[16 * PITCH];
  __shared__ __align__(16) unsigned int sLo[16 * PITCH];

  int tid = threadIdx.x;
  int row0 = blockIdx.x * 16;

  for (int i = tid; i < 16 * (DDIM / 2); i += 256) {
    int r = i / (DDIM / 2), off = i % (DDIM / 2);
    int node = row0 + r;
    if (node > N - 1) node = N - 1;
    float2 f = *(const float2*)(x + (size_t)node * DDIM + off * 2);
    unsigned int h, l;
    split_pack(f.x, f.y, h, l);
    sHi[r * PITCH + off] = h;
    sLo[r * PITCH + off] = l;
  }
  __syncthreads();

  int wave = tid >> 5, lane = tid & 31, lg = lane >> 4, n = lane & 15;
  int nt0 = wave * 2;
  v8f c0 = {0.f, 0.f, 0.f, 0.f, 0.f, 0.f, 0.f, 0.f};
  v8f c1 = {0.f, 0.f, 0.f, 0.f, 0.f, 0.f, 0.f, 0.f};
  const uint4* rowH = (const uint4*)(sHi + n * PITCH);
  const uint4* rowL = (const uint4*)(sLo + n * PITCH);

  for (int kt = 0; kt < KT; ++kt) {
    BFrag aH, aL, b0H, b0L, b1H, b1L;
    aH.q[0] = rowH[kt * 4 + lg];     aH.q[1] = rowH[kt * 4 + 2 + lg];
    aL.q[0] = rowL[kt * 4 + lg];     aL.q[1] = rowL[kt * 4 + 2 + lg];
    size_t b0 = ((size_t)(kt * 16 + nt0) * 32 + lane) * 8;
    size_t b1 = b0 + 256;  // next ntile
    b0H.q[0] = *(const uint4*)(WHi + b0); b0H.q[1] = *(const uint4*)(WHi + b0 + 4);
    b0L.q[0] = *(const uint4*)(WLo + b0); b0L.q[1] = *(const uint4*)(WLo + b0 + 4);
    b1H.q[0] = *(const uint4*)(WHi + b1); b1H.q[1] = *(const uint4*)(WHi + b1 + 4);
    b1L.q[0] = *(const uint4*)(WLo + b1); b1L.q[1] = *(const uint4*)(WLo + b1 + 4);
    c0 = wmma_bf16(aH, b0H, c0); c0 = wmma_bf16(aH, b0L, c0); c0 = wmma_bf16(aL, b0H, c0);
    c1 = wmma_bf16(aH, b1H, c1); c1 = wmma_bf16(aH, b1L, c1); c1 = wmma_bf16(aL, b1H, c1);
  }

  int col0 = nt0 * 16 + n, col1 = col0 + 16;
  float bias0 = bc[col0], bias1 = bc[col1];
#pragma unroll
  for (int v = 0; v < 8; ++v) {
    int row = row0 + v + 8 * lg;
    if (row < N) {
      out[(size_t)row * DDIM + col0] = c0[v] + bias0;
      out[(size_t)row * DDIM + col1] = c1[v] + bias1;
    }
  }
}

// ---------------------------------------------------------------------------
// Edge GEMM: 16 same-type edges per block; gather -> WMMA -> normalized
// atomic scatter-add into out.
// ---------------------------------------------------------------------------
template <int SHAPE>
__global__ __launch_bounds__(256) void hgnn_edge_wmma(
    const float* __restrict__ x, const int* __restrict__ idx,
    const int* __restrict__ counts, const int* __restrict__ lens,
    const int* __restrict__ lists, const unsigned int* __restrict__ BHi,
    const unsigned int* __restrict__ BLo, float* __restrict__ out,
    int E, int N, int tilesPerType) {
  constexpr int KDIM = SHAPE * DDIM;
  constexpr int KT = KDIM / 32;
  constexpr int PITCH = KDIM / 2 + 4;  // dwords
  __shared__ __align__(16) unsigned int sHi[16 * PITCH];
  __shared__ __align__(16) unsigned int sLo[16 * PITCH];
  __shared__ int   sSrc[16 * SHAPE];
  __shared__ int   sDst[16];
  __shared__ float sNorm[16];

  int t    = blockIdx.x / tilesPerType;
  int tile = blockIdx.x % tilesPerType;
  int len  = lens[t];
  int base = tile * 16;
  if (base >= len) return;  // uniform across block

  int tid = threadIdx.x;
  if (tid < 16) {
    int slot = base + tid;
    int dst = 0, s0 = 0, s1 = 0;
    float norm = 0.f;  // norm==0 marks an invalid (padding) slot
    if (slot < len) {
      int e = lists[(size_t)t * E + slot];
      s0 = idx[(size_t)SHAPE * e];
      if (SHAPE == 2) s1 = idx[(size_t)SHAPE * e + 1];
      dst = idx[(size_t)E * SHAPE + (size_t)SHAPE * e];
      int c = counts[(size_t)t * N + dst];
      norm = 1.0f / (float)(c > 1 ? c : 1);
    }
    sSrc[SHAPE * tid] = s0;
    if (SHAPE == 2) sSrc[SHAPE * tid + 1] = s1;
    sDst[tid] = dst;
    sNorm[tid] = norm;
  }
  __syncthreads();

  // gather: 16 rows of K fp32 features, split into bf16 hi/lo planes in LDS
  for (int i = tid; i < 16 * (KDIM / 2); i += 256) {
    int r = i / (KDIM / 2), off = i % (KDIM / 2);
    int foff = off * 2;
    int node;
    if (SHAPE == 1) {
      node = sSrc[r];
    } else {
      int half = foff >> 8;          // which concatenated source node
      node = sSrc[2 * r + half];
      foff &= 255;
    }
    float2 f = *(const float2*)(x + (size_t)node * DDIM + foff);
    unsigned int h, l;
    split_pack(f.x, f.y, h, l);
    sHi[r * PITCH + off] = h;
    sLo[r * PITCH + off] = l;
  }
  __syncthreads();

  int wave = tid >> 5, lane = tid & 31, lg = lane >> 4, n = lane & 15;
  int nt0 = wave * 2;
  v8f c0 = {0.f, 0.f, 0.f, 0.f, 0.f, 0.f, 0.f, 0.f};
  v8f c1 = {0.f, 0.f, 0.f, 0.f, 0.f, 0.f, 0.f, 0.f};
  const uint4* rowH = (const uint4*)(sHi + n * PITCH);
  const uint4* rowL = (const uint4*)(sLo + n * PITCH);
  const unsigned int* BHt = BHi + (size_t)t * KT * 16 * 32 * 8;
  const unsigned int* BLt = BLo + (size_t)t * KT * 16 * 32 * 8;

  for (int kt = 0; kt < KT; ++kt) {
    BFrag aH, aL, b0H, b0L, b1H, b1L;
    aH.q[0] = rowH[kt * 4 + lg];     aH.q[1] = rowH[kt * 4 + 2 + lg];
    aL.q[0] = rowL[kt * 4 + lg];     aL.q[1] = rowL[kt * 4 + 2 + lg];
    size_t b0 = ((size_t)(kt * 16 + nt0) * 32 + lane) * 8;
    size_t b1 = b0 + 256;
    b0H.q[0] = *(const uint4*)(BHt + b0); b0H.q[1] = *(const uint4*)(BHt + b0 + 4);
    b0L.q[0] = *(const uint4*)(BLt + b0); b0L.q[1] = *(const uint4*)(BLt + b0 + 4);
    b1H.q[0] = *(const uint4*)(BHt + b1); b1H.q[1] = *(const uint4*)(BHt + b1 + 4);
    b1L.q[0] = *(const uint4*)(BLt + b1); b1L.q[1] = *(const uint4*)(BLt + b1 + 4);
    c0 = wmma_bf16(aH, b0H, c0); c0 = wmma_bf16(aH, b0L, c0); c0 = wmma_bf16(aL, b0H, c0);
    c1 = wmma_bf16(aH, b1H, c1); c1 = wmma_bf16(aH, b1L, c1); c1 = wmma_bf16(aL, b1H, c1);
  }

  int col0 = nt0 * 16 + n, col1 = col0 + 16;
#pragma unroll
  for (int v = 0; v < 8; ++v) {
    int m = v + 8 * lg;
    float nm = sNorm[m];
    if (nm != 0.f) {
      size_t o = (size_t)sDst[m] * DDIM;
      atomic_fadd(out + o + col0, nm * c0[v]);
      atomic_fadd(out + o + col1, nm * c1[v]);
    }
  }
}

// ---------------------------------------------------------------------------
extern "C" void kernel_launch(void* const* d_in, const int* in_sizes, int n_in,
                              void* d_out, int out_size, void* d_ws, size_t ws_size,
                              hipStream_t stream) {
  (void)n_in; (void)out_size; (void)ws_size;
  const float* x  = (const float*)d_in[0];
  const int* hi1  = (const int*)d_in[1];
  const int* ht1  = (const int*)d_in[2];
  const int* hi2  = (const int*)d_in[3];
  const int* ht2  = (const int*)d_in[4];
  const float* A1 = (const float*)d_in[5];
  const float* A2 = (const float*)d_in[6];
  const float* Wc = (const float*)d_in[7];
  const float* bc = (const float*)d_in[8];
  float* out = (float*)d_out;

  const int N  = in_sizes[0] / DDIM;          // 100000
  const int E1 = in_sizes[2];                 // 300000
  const int E2 = in_sizes[4];                 // 300000
  const int T  = in_sizes[5] / (DDIM * DDIM); // 4

  // workspace carve-up (256B aligned)
  char* ws = (char*)d_ws;
  size_t off = 0;
  auto carve = [&](size_t bytes) -> void* {
    void* p = ws + off;
    off = (off + bytes + 255) & ~((size_t)255);
    return p;
  };
  const size_t b1dw = (size_t)T * 8 * 16 * 32 * 8;   // swizzled dwords, K=256
  const size_t b2dw = (size_t)T * 16 * 16 * 32 * 8;  // K=512
  const size_t wdw  = (size_t)1 * 8 * 16 * 32 * 8;
  unsigned int* B1h = (unsigned int*)carve(b1dw * 4);
  unsigned int* B1l = (unsigned int*)carve(b1dw * 4);
  unsigned int* B2h = (unsigned int*)carve(b2dw * 4);
  unsigned int* B2l = (unsigned int*)carve(b2dw * 4);
  unsigned int* Wh  = (unsigned int*)carve(wdw * 4);
  unsigned int* Wl  = (unsigned int*)carve(wdw * 4);
  size_t zero_begin = off;
  int* cnt1 = (int*)carve((size_t)T * N * 4);
  int* cnt2 = (int*)carve((size_t)T * N * 4);
  int* lens = (int*)carve((size_t)2 * T * 4);  // [0..T): shape1, [T..2T): shape2
  size_t zero_end = off;
  int* list1 = (int*)carve((size_t)T * E1 * 4);
  int* list2 = (int*)carve((size_t)T * E2 * 4);

  hipMemsetAsync(ws + zero_begin, 0, zero_end - zero_begin, stream);

  // 1) swizzle weights into WMMA-fragment order (hi/lo bf16 planes)
  {
    int tot1 = T * 8 * 16 * 32 * 8;
    int tot2 = T * 16 * 16 * 32 * 8;
    int totw = 8 * 16 * 32 * 8;
    hgnn_swizzleB<256, false><<<(tot1 + 255) / 256, 256, 0, stream>>>(A1, B1h, B1l, T);
    hgnn_swizzleB<512, false><<<(tot2 + 255) / 256, 256, 0, stream>>>(A2, B2h, B2l, T);
    hgnn_swizzleB<256, true ><<<(totw + 255) / 256, 256, 0, stream>>>(Wc, Wh, Wl, 1);
  }

  // 2) per-(type,dst) counts + type buckets
  hgnn_count_bucket<<<(E1 + 255) / 256, 256, 0, stream>>>(hi1, ht1, cnt1, lens,     list1, E1, N, 1);
  hgnn_count_bucket<<<(E2 + 255) / 256, 256, 0, stream>>>(hi2, ht2, cnt2, lens + T, list2, E2, N, 2);

  // 3) dense term writes out (initializes it), then edge terms accumulate
  hgnn_dense_wmma<<<(N + 15) / 16, 256, 0, stream>>>(x, Wh, Wl, bc, out, N);

  const int tpt1 = (E1 + 15) / 16;
  const int tpt2 = (E2 + 15) / 16;
  hgnn_edge_wmma<1><<<T * tpt1, 256, 0, stream>>>(x, hi1, cnt1, lens,     list1, B1h, B1l, out, E1, N, tpt1);
  hgnn_edge_wmma<2><<<T * tpt2, 256, 0, stream>>>(x, hi2, cnt2, lens + T, list2, B2h, B2l, out, E2, N, tpt2);
}
